// GPT_68805376082021
// MI455X (gfx1250) — matrix operand order repaired and checked by reading, not compiled, not run on hardware
//
#include <hip/hip_runtime.h>
#include <hip/hip_bf16.h>
#include <math.h>

#define LNUM 4
#define BB   4
#define TT   2048
#define CC   512
#define HH   8
#define DD   64
#define FEC  2048   /* FE*C */
#define BT   8192   /* B*T  */

typedef __attribute__((ext_vector_type(16))) __bf16 v16bf;
typedef __attribute__((ext_vector_type(8)))  float  v8f;

struct __align__(16) U4 { unsigned int x, y, z, w; };
union Frag { U4 u[2]; v16bf v; };

__device__ __forceinline__ unsigned short f2bf(float f) {
  unsigned int u = __float_as_uint(f);
  u = (u + 0x7FFFu + ((u >> 16) & 1u)) >> 16;   // round-to-nearest-even
  return (unsigned short)u;
}

__device__ __forceinline__ float geluf(float x) {
  return 0.5f * x * (1.0f + erff(x * 0.70710678118654752f));
}

// ---- CDNA5 async global->LDS copy (ASYNCcnt path, CDNA5 ISA) ----
// Copies 16B per lane from the lane's global address into the lane's LDS offset.
__device__ __forceinline__ void async_cp16(unsigned lds_off, const void* gaddr) {
  asm volatile("global_load_async_to_lds_b128 %0, %1, off"
               :: "v"(lds_off), "v"(gaddr) : "memory");
}
template<int N>
__device__ __forceinline__ void wait_asynccnt() {
  asm volatile("s_wait_asynccnt %0" :: "n"(N) : "memory");
}
__device__ __forceinline__ unsigned lds_off32(const void* p) {
  // generic shared address = {aperture_hi, lds_byte_offset}; low 32 bits = LDS offset
  return (unsigned)(unsigned long long)p;
}

// ---------------- LayerNorm: one wave per 512-wide row ----------------
template<int OUT_BF16>
__global__ __launch_bounds__(256) void ln_kernel(const float* __restrict__ x,
    const float* __restrict__ w, const float* __restrict__ b,
    void* __restrict__ out)
{
  const int lane = threadIdx.x & 31;
  const int row  = blockIdx.x * 8 + (threadIdx.x >> 5);
  const float* xr = x + (size_t)row * CC;
  float v[16]; float s = 0.f;
#pragma unroll
  for (int i = 0; i < 16; ++i) { v[i] = xr[lane + i * 32]; s += v[i]; }
#pragma unroll
  for (int off = 16; off >= 1; off >>= 1) s += __shfl_xor(s, off, 32);
  const float mu = s * (1.0f / CC);
  float var = 0.f;
#pragma unroll
  for (int i = 0; i < 16; ++i) { float d = v[i] - mu; var += d * d; }
#pragma unroll
  for (int off = 16; off >= 1; off >>= 1) var += __shfl_xor(var, off, 32);
  const float rs = rsqrtf(var * (1.0f / CC) + 1e-5f);
#pragma unroll
  for (int i = 0; i < 16; ++i) {
    const int c = lane + i * 32;
    const float o = (v[i] - mu) * rs * w[c] + b[c];
    if (OUT_BF16) ((unsigned short*)out)[(size_t)row * CC + c] = f2bf(o);
    else          ((float*)out)[(size_t)row * CC + c] = o;
  }
}

// ------- fp32 [K,N] weight -> bf16 transposed [N,K], one z-layer per blockIdx.y -------
__global__ __launch_bounds__(256) void cvtT_kernel(const float* __restrict__ in,
    unsigned short* __restrict__ out, int Kd, int Nd)
{
  const size_t total = (size_t)Kd * Nd;
  const size_t base  = (size_t)blockIdx.y * total;
  for (size_t i = (size_t)blockIdx.x * 256 + threadIdx.x; i < total;
       i += (size_t)gridDim.x * 256) {
    const size_t kk = i / (size_t)Nd, nn = i % (size_t)Nd;
    out[base + nn * (size_t)Kd + kk] = f2bf(in[base + i]);
  }
}

// ---------------- WMMA GEMM: C[M,N] = A[M,K] * Bt[N,K]^T + bias ----------------
// 256 threads = 8 waves (2x4). Block tile 128x128, K-step 32. Wave tile 64x32.
// Double-buffered LDS fed by global_load_async_to_lds_b128 (4 per thread per tile);
// s_wait_asynccnt 4 retires the previous tile while the next one streams in.
// MODE 0: store bf16; MODE 1: gelu -> bf16; MODE 2: f32 residual add (Out = Resid + val)
template<int MODE>
__global__ __launch_bounds__(256) void gemm_kernel(
    const unsigned short* __restrict__ A, const unsigned short* __restrict__ Bt,
    const float* __restrict__ bias, void* __restrict__ Out,
    const float* __restrict__ Resid, int N, int K)
{
  __shared__ __align__(16) unsigned short As[2][128 * 40];  // row stride 80B (16B-aligned, conflict-free)
  __shared__ __align__(16) unsigned short Bs[2][128 * 40];
  const int tid  = threadIdx.x;
  const int lane = tid & 31, wid = tid >> 5;
  const int wm = wid & 1, wn = wid >> 1;
  const int hi = lane >> 4, nl = lane & 15;
  const int bm = blockIdx.y * 128, bn = blockIdx.x * 128;

  // per-thread staging coordinates (2 x 16B chunks of A and of B per tile)
  const int c0 = tid,        r0 = c0 >> 2, e0 = (c0 & 3) * 8;
  const int c1 = tid + 256,  r1 = c1 >> 2, e1 = (c1 & 3) * 8;
  const unsigned short* gA0 = A  + (size_t)(bm + r0) * K + e0;
  const unsigned short* gA1 = A  + (size_t)(bm + r1) * K + e1;
  const unsigned short* gB0 = Bt + (size_t)(bn + r0) * K + e0;
  const unsigned short* gB1 = Bt + (size_t)(bn + r1) * K + e1;
  const int lA0 = r0 * 40 + e0, lA1 = r1 * 40 + e1;

  auto stage = [&](int buf, int k0) {
    async_cp16(lds_off32(&As[buf][lA0]), gA0 + k0);
    async_cp16(lds_off32(&As[buf][lA1]), gA1 + k0);
    async_cp16(lds_off32(&Bs[buf][lA0]), gB0 + k0);
    async_cp16(lds_off32(&Bs[buf][lA1]), gB1 + k0);
  };

  v8f acc[4][2] = {};
  const int nk = K >> 5;
  stage(0, 0);
  for (int t = 0; t < nk; ++t) {
    if (t + 1 < nk) { stage((t + 1) & 1, (t + 1) * 32); wait_asynccnt<4>(); }
    else            { wait_asynccnt<0>(); }
    __syncthreads();                       // tile t fully in LDS for all waves
    const unsigned short* as = As[t & 1];
    const unsigned short* bs = Bs[t & 1];
#pragma unroll
    for (int mt = 0; mt < 4; ++mt) {
      Frag fra;
      const int ra = (wm * 64 + mt * 16 + nl) * 40;
      fra.u[0] = *(const U4*)&as[ra + hi * 8];        // K 0..7 / 8..15
      fra.u[1] = *(const U4*)&as[ra + hi * 8 + 16];   // K 16..23 / 24..31
#pragma unroll
      for (int nt = 0; nt < 2; ++nt) {
        Frag frb;
        const int rb = (wn * 32 + nt * 16 + nl) * 40;
        frb.u[0] = *(const U4*)&bs[rb + hi * 8];      // K 0..7 / 8..15
        frb.u[1] = *(const U4*)&bs[rb + hi * 8 + 16]; // K 16..23 / 24..31
        acc[mt][nt] = __builtin_amdgcn_wmma_f32_16x16x32_bf16(
            false, fra.v, false, frb.v, (short)0, acc[mt][nt], false, false);
      }
    }
    __syncthreads();                       // done reading before buffer reuse
  }
#pragma unroll
  for (int mt = 0; mt < 4; ++mt)
#pragma unroll
    for (int nt = 0; nt < 2; ++nt)
#pragma unroll
      for (int r = 0; r < 8; ++r) {
        const int row = bm + wm * 64 + mt * 16 + r + 8 * hi;
        const int col = bn + wn * 32 + nt * 16 + nl;
        float vv = acc[mt][nt][r] + bias[col];
        if (MODE == 1) vv = geluf(vv);
        const size_t idx = (size_t)row * N + col;
        if (MODE == 2) ((float*)Out)[idx] = Resid[idx] + vv;
        else           ((unsigned short*)Out)[idx] = f2bf(vv);
      }
}

// ---------------- Flash attention (causal), bf16 WMMA, online softmax ----------------
// grid (T/64, H, B), 128 threads = 4 waves; each wave owns 16 query rows.
__global__ __launch_bounds__(128) void fa_kernel(const unsigned short* __restrict__ Q,
    const unsigned short* __restrict__ Kp, const unsigned short* __restrict__ Vp,
    unsigned short* __restrict__ Y)
{
  __shared__ __align__(16) unsigned short Ks[64 * 72];  // [key][d], stride 144B
  __shared__ __align__(16) unsigned short Vt[64 * 72];  // [d][key] (transposed)
  __shared__ __align__(16) unsigned short Pl[64 * 72];  // wave-private P tiles
  const int tid = threadIdx.x, lane = tid & 31, wid = tid >> 5;
  const int hi = lane >> 4, nl = lane & 15;
  const int b = blockIdx.z, h = blockIdx.y, qb = blockIdx.x;
  const int qRow = qb * 64 + wid * 16 + nl;
  const unsigned short* qp = Q + (size_t)(b * TT + qRow) * CC + h * DD;
  Frag fq[2];
#pragma unroll
  for (int kc = 0; kc < 2; ++kc) {
    fq[kc].u[0] = *(const U4*)(qp + kc * 32 + hi * 8);
    fq[kc].u[1] = *(const U4*)(qp + kc * 32 + hi * 8 + 16);
  }
  float mrun[8], lrun[8];
  v8f Ov[4] = {};
#pragma unroll
  for (int r = 0; r < 8; ++r) { mrun[r] = -1e30f; lrun[r] = 0.f; }

  for (int kb = 0; kb <= qb; ++kb) {            // causal: only lower-tri key blocks
    const int kBase = kb * 64;
    for (int c = tid; c < 512; c += 128) {      // stage K block (async, no VGPR round-trip)
      const int r = c >> 3, c8 = c & 7;
      async_cp16(lds_off32(&Ks[r * 72 + c8 * 8]),
                 Kp + (size_t)(b * TT + kBase + r) * CC + h * DD + c8 * 8);
    }
    for (int e = tid; e < 4096; e += 128) {     // stage V transposed
      const int key = e >> 6, d = e & 63;
      Vt[d * 72 + key] = Vp[(size_t)(b * TT + kBase + key) * CC + h * DD + d];
    }
    wait_asynccnt<0>();
    __syncthreads();

    float sv[4][8];
#pragma unroll
    for (int nt = 0; nt < 4; ++nt) {            // S = Q * K^T (K-dim = D = 64)
      v8f a = {};
#pragma unroll
      for (int kc = 0; kc < 2; ++kc) {
        Frag frb;
        const int rb = (nt * 16 + nl) * 72 + kc * 32;
        frb.u[0] = *(const U4*)&Ks[rb + hi * 8];
        frb.u[1] = *(const U4*)&Ks[rb + hi * 8 + 16];
        a = __builtin_amdgcn_wmma_f32_16x16x32_bf16(
            false, fq[kc].v, false, frb.v, (short)0, a, false, false);
      }
#pragma unroll
      for (int r = 0; r < 8; ++r) {
        const int rowg = qb * 64 + wid * 16 + r + 8 * hi;
        const int colg = kBase + nt * 16 + nl;
        sv[nt][r] = (colg <= rowg) ? a[r] * 0.125f : -1e30f;
      }
    }
    // online softmax per query row (16-lane cross reductions stay in half-wave)
#pragma unroll
    for (int r = 0; r < 8; ++r) {
      float m = fmaxf(fmaxf(sv[0][r], sv[1][r]), fmaxf(sv[2][r], sv[3][r]));
#pragma unroll
      for (int off = 8; off >= 1; off >>= 1) m = fmaxf(m, __shfl_xor(m, off, 32));
      const float mnew  = fmaxf(mrun[r], m);
      const float alpha = __expf(mrun[r] - mnew);
      mrun[r] = mnew;
      float rs = 0.f;
#pragma unroll
      for (int nt = 0; nt < 4; ++nt) { sv[nt][r] = __expf(sv[nt][r] - mnew); rs += sv[nt][r]; }
#pragma unroll
      for (int off = 8; off >= 1; off >>= 1) rs += __shfl_xor(rs, off, 32);
      lrun[r] = lrun[r] * alpha + rs;
#pragma unroll
      for (int j = 0; j < 4; ++j) Ov[j][r] *= alpha;
    }
    // reshape P into WMMA A-layout through wave-private LDS
#pragma unroll
    for (int nt = 0; nt < 4; ++nt)
#pragma unroll
      for (int r = 0; r < 8; ++r)
        Pl[(wid * 16 + r + 8 * hi) * 72 + nt * 16 + nl] = f2bf(sv[nt][r]);
    asm volatile("s_wait_dscnt 0x0" ::: "memory");
    // O += P * V   (K-dim = 64 keys)
#pragma unroll
    for (int j = 0; j < 4; ++j) {
#pragma unroll
      for (int kc = 0; kc < 2; ++kc) {
        Frag fra, frb;
        const int rp = (wid * 16 + nl) * 72 + kc * 32;
        fra.u[0] = *(const U4*)&Pl[rp + hi * 8];
        fra.u[1] = *(const U4*)&Pl[rp + hi * 8 + 16];
        const int rv = (j * 16 + nl) * 72 + kc * 32;
        frb.u[0] = *(const U4*)&Vt[rv + hi * 8];
        frb.u[1] = *(const U4*)&Vt[rv + hi * 8 + 16];
        Ov[j] = __builtin_amdgcn_wmma_f32_16x16x32_bf16(
            false, fra.v, false, frb.v, (short)0, Ov[j], false, false);
      }
    }
    __syncthreads();
  }
#pragma unroll
  for (int r = 0; r < 8; ++r) lrun[r] = 1.0f / lrun[r];
#pragma unroll
  for (int j = 0; j < 4; ++j)
#pragma unroll
    for (int r = 0; r < 8; ++r) {
      const int row = qb * 64 + wid * 16 + r + 8 * hi;
      Y[(size_t)(b * TT + row) * CC + h * DD + j * 16 + nl] = f2bf(Ov[j][r] * lrun[r]);
    }
}

// ---------------------------------------------------------------------------
extern "C" void kernel_launch(void* const* d_in, const int* in_sizes, int n_in,
                              void* d_out, int out_size, void* d_ws, size_t ws_size,
                              hipStream_t stream) {
  (void)in_sizes; (void)n_in; (void)out_size; (void)ws_size;
  const float* x    = (const float*)d_in[0];
  const float* wq   = (const float*)d_in[1];
  const float* bq   = (const float*)d_in[2];
  const float* wk   = (const float*)d_in[3];
  const float* bk   = (const float*)d_in[4];
  const float* wv   = (const float*)d_in[5];
  const float* bv   = (const float*)d_in[6];
  const float* wo   = (const float*)d_in[7];
  const float* bo   = (const float*)d_in[8];
  const float* ln1w = (const float*)d_in[9];
  const float* ln1b = (const float*)d_in[10];
  const float* ln2w = (const float*)d_in[11];
  const float* ln2b = (const float*)d_in[12];
  const float* w1   = (const float*)d_in[13];
  const float* b1   = (const float*)d_in[14];
  const float* w2   = (const float*)d_in[15];
  const float* b2   = (const float*)d_in[16];
  const float* lnfw = (const float*)d_in[17];
  const float* lnfb = (const float*)d_in[18];

  char* p = (char*)d_ws;
  auto take = [&](size_t bytes) { char* r = p; p += (bytes + 255) & ~(size_t)255; return r; };
  float*          xf  = (float*)take((size_t)BT * CC * 4);            // residual stream
  unsigned short* hln = (unsigned short*)take((size_t)BT * CC * 2);   // LN output (bf16)
  unsigned short* qb_ = (unsigned short*)take((size_t)BT * CC * 2);
  unsigned short* kb_ = (unsigned short*)take((size_t)BT * CC * 2);
  unsigned short* vb_ = (unsigned short*)take((size_t)BT * CC * 2);
  unsigned short* yb_ = (unsigned short*)take((size_t)BT * CC * 2);
  unsigned short* h1b = (unsigned short*)take((size_t)BT * FEC * 2);
  unsigned short* wqb = (unsigned short*)take((size_t)LNUM * CC * CC * 2);
  unsigned short* wkb = (unsigned short*)take((size_t)LNUM * CC * CC * 2);
  unsigned short* wvb = (unsigned short*)take((size_t)LNUM * CC * CC * 2);
  unsigned short* wob = (unsigned short*)take((size_t)LNUM * CC * CC * 2);
  unsigned short* w1b = (unsigned short*)take((size_t)LNUM * CC * FEC * 2);
  unsigned short* w2b = (unsigned short*)take((size_t)LNUM * FEC * CC * 2);

  hipMemcpyAsync(xf, x, (size_t)BT * CC * 4, hipMemcpyDeviceToDevice, stream);

  dim3 cg(1024, LNUM);
  cvtT_kernel<<<cg, 256, 0, stream>>>(wq, wqb, CC, CC);
  cvtT_kernel<<<cg, 256, 0, stream>>>(wk, wkb, CC, CC);
  cvtT_kernel<<<cg, 256, 0, stream>>>(wv, wvb, CC, CC);
  cvtT_kernel<<<cg, 256, 0, stream>>>(wo, wob, CC, CC);
  cvtT_kernel<<<cg, 256, 0, stream>>>(w1, w1b, CC, FEC);
  cvtT_kernel<<<cg, 256, 0, stream>>>(w2, w2b, FEC, CC);

  const dim3 gC(CC / 128, BT / 128);   // 4 x 64
  const dim3 gF(FEC / 128, BT / 128);  // 16 x 64
  const dim3 gA(TT / 64, HH, BB);      // 32 x 8 x 4

  for (int l = 0; l < LNUM; ++l) {
    ln_kernel<1><<<BT / 8, 256, 0, stream>>>(xf, ln1w + l * CC, ln1b + l * CC, hln);
    gemm_kernel<0><<<gC, 256, 0, stream>>>(hln, wqb + (size_t)l * CC * CC, bq + l * CC, qb_, nullptr, CC, CC);
    gemm_kernel<0><<<gC, 256, 0, stream>>>(hln, wkb + (size_t)l * CC * CC, bk + l * CC, kb_, nullptr, CC, CC);
    gemm_kernel<0><<<gC, 256, 0, stream>>>(hln, wvb + (size_t)l * CC * CC, bv + l * CC, vb_, nullptr, CC, CC);
    fa_kernel<<<gA, 128, 0, stream>>>(qb_, kb_, vb_, yb_);
    gemm_kernel<2><<<gC, 256, 0, stream>>>(yb_, wob + (size_t)l * CC * CC, bo + l * CC, xf, xf, CC, CC);
    ln_kernel<1><<<BT / 8, 256, 0, stream>>>(xf, ln2w + l * CC, ln2b + l * CC, hln);
    gemm_kernel<1><<<gF, 256, 0, stream>>>(hln, w1b + (size_t)l * CC * FEC, b1 + l * FEC, h1b, nullptr, FEC, CC);
    gemm_kernel<2><<<gC, 256, 0, stream>>>(h1b, w2b + (size_t)l * FEC * CC, b2 + l * CC, xf, xf, CC, FEC);
  }
  ln_kernel<0><<<BT / 8, 256, 0, stream>>>(xf, lnfw, lnfb, d_out);
}
